// ProposalModule_35691178230475
// MI455X (gfx1250) — matrix operand order repaired and checked by reading, not compiled, not run on hardware
//
#include <hip/hip_runtime.h>
#include <hip/hip_bf16.h>

#define B_      8
#define K_      8192
#define C_      256
#define NPROP   512
#define NSAMP   16
#define RADIUS_ 0.3f
#define K1PAD   288      // 259 -> padded to multiple of 32
#define KB1     9        // k-blocks of 32 for layer 1
#define KBC     4        // k-blocks of 32 for 128-channel layers
#define NCH     128
#define OUTC    119
#define NCOL    (B_*NPROP)   // 4096 proposal columns for FC stage

typedef __attribute__((ext_vector_type(16))) __bf16   v16bf;
typedef __attribute__((ext_vector_type(8)))  float    v8f;
typedef __attribute__((ext_vector_type(8)))  unsigned v8u;
typedef __attribute__((ext_vector_type(4)))  unsigned v4u;

__device__ __forceinline__ unsigned short f2bf(float x) {
    unsigned u = __builtin_bit_cast(unsigned, x);
    unsigned r = 0x7FFFu + ((u >> 16) & 1u);       // round-to-nearest-even
    return (unsigned short)((u + r) >> 16);
}

// ---------------------------------------------------------------------------
// Operand-layout helpers (CDNA5 ISA 7.12.2, wave32).
//
// A operand stored pre-swizzled in global: [mtile][kblock][lane][16 bf16],
// the 16 values in VGPR-pair order, so a lane's A tile is one 32B load.
__device__ __forceinline__ v16bf load_a(const unsigned short* Wsw, int KB,
                                        int mt, int kk, int lane) {
    const v8u* p = (const v8u*)(Wsw + (((mt * KB + kk) * 32 + lane) << 4));
    return __builtin_bit_cast(v16bf, *p);
}

// B operand stored swizzled in LDS: [kblock][lane][16 bf16] with
// lane = col + 16*k_half; a lane's B tile is one 32B LDS load.
__device__ __forceinline__ v16bf load_b(const unsigned short* H, int kk, int lane) {
    const v8u* p = (const v8u*)(H + (((kk * 32) + lane) << 4));
    return __builtin_bit_cast(v16bf, *p);
}

// Swizzled LDS element index for activation element (channel ch, column col).
__device__ __forceinline__ int swz(int ch, int col) {
    return ((ch >> 5) << 9) + ((col + (((ch >> 4) & 1) << 4)) << 4) + (ch & 15);
}

// Apply folded BN (+ReLU) to a 16x16 D tile and write to swizzled LDS with a
// single 16B store per thread (channels w*16+sub*8+j are contiguous there).
__device__ __forceinline__ void bn_relu_store(v8f acc, int layer, int w, int sub, int rr,
                                              const float* scale, const float* shift,
                                              unsigned short* H) {
    v4u pk;
#pragma unroll
    for (int d = 0; d < 4; ++d) {
        int ch0 = w * 16 + sub * 8 + 2 * d;
        float y0 = fmaxf(acc[2 * d]     * scale[layer * NCH + ch0]     + shift[layer * NCH + ch0],     0.0f);
        float y1 = fmaxf(acc[2 * d + 1] * scale[layer * NCH + ch0 + 1] + shift[layer * NCH + ch0 + 1], 0.0f);
        pk[d] = (unsigned)f2bf(y0) | ((unsigned)f2bf(y1) << 16);
    }
    int base = ((w >> 1) << 9) + ((rr + ((w & 1) << 4)) << 4) + sub * 8;
    *(v4u*)(H + base) = pk;     // ds_store_b128
}

// ---------------------------------------------------------------------------
// Kernel 0: weight conversion to bf16 pre-swizzled into A-operand order + BN fold.
__device__ __forceinline__ void swz_weight(const float* src, unsigned short* dst,
                                           int i, int Korig, int Mrows, int KB) {
    int q = i & 15, lane = (i >> 4) & 31;
    int blk = i >> 9;
    int kk = blk % KB, mt = blk / KB;
    int rr = lane & 15, sub = lane >> 4;
    int v = q >> 1, par = q & 1;
    int kofs = 16 * sub + q;
    int row = mt * 16 + rr, k = kk * 32 + kofs;
    float val = (row < Mrows && k < Korig) ? src[row * Korig + k] : 0.0f;
    dst[i] = f2bf(val);
}

__global__ void prep_kernel(const float* w1, const float* w2, const float* w3,
                            const float* wc1, const float* bc1, const float* wc2,
                            const float* bc2, const float* wc3, const float* bc3,
                            const float* bng, const float* bnb, const float* bnm,
                            const float* bnv,
                            unsigned short* W1s, unsigned short* W2s, unsigned short* W3s,
                            unsigned short* WC1s, unsigned short* WC2s, unsigned short* WC3s,
                            float* scale, float* shift, float* bc3f) {
    const int T1 = 8 * KB1 * 512;            // 36864
    const int TC = 8 * KBC * 512;            // 16384
    const int total = T1 + 5 * TC + 5 * 128 + 128;
    int stride = gridDim.x * blockDim.x;
    for (int i = blockIdx.x * blockDim.x + threadIdx.x; i < total; i += stride) {
        if (i < T1) { swz_weight(w1, W1s, i, 259, 128, KB1); }
        else {
            int j = i - T1;
            if      (j < TC)      swz_weight(w2,  W2s,  j,          128, 128, KBC);
            else if (j < 2 * TC)  swz_weight(w3,  W3s,  j - TC,     128, 128, KBC);
            else if (j < 3 * TC)  swz_weight(wc1, WC1s, j - 2 * TC, 128, 128, KBC);
            else if (j < 4 * TC)  swz_weight(wc2, WC2s, j - 3 * TC, 128, 128, KBC);
            else if (j < 5 * TC)  swz_weight(wc3, WC3s, j - 4 * TC, 128, OUTC, KBC);
            else if (j < 5 * TC + 640) {
                int k = j - 5 * TC; int layer = k / 128, ch = k % 128;
                float sc = bng[k] * rsqrtf(bnv[k] + 1e-5f);
                float sh = bnb[k] - bnm[k] * sc;
                if (layer == 3) sh += bc1[ch] * sc;   // bias-before-BN fold
                if (layer == 4) sh += bc2[ch] * sc;
                scale[k] = sc; shift[k] = sh;
            } else {
                int ch = j - 5 * TC - 640;
                bc3f[ch] = (ch < OUTC) ? bc3[ch] : 0.0f;
            }
        }
    }
}

// ---------------------------------------------------------------------------
// Kernel 1: furthest point sampling. One workgroup per batch; dist[] in LDS.
__global__ __launch_bounds__(1024) void fps_kernel(const float* __restrict__ xyz,
                                                   int* fpsidx, float* newxyz) {
    int b = blockIdx.x;
    const float* X = xyz + (size_t)b * K_ * 3;
    __shared__ float dist[K_];
    __shared__ float rv[32];
    __shared__ int   ri[32];
    __shared__ int   bidx;
    int t = threadIdx.x, lane = t & 31, w = t >> 5;
    for (int p = t; p < K_; p += 1024) dist[p] = 1e10f;
    float lx = X[0], ly = X[1], lz = X[2];
    if (t == 0) {
        fpsidx[b * NPROP] = 0;
        newxyz[(b * NPROP) * 3 + 0] = lx;
        newxyz[(b * NPROP) * 3 + 1] = ly;
        newxyz[(b * NPROP) * 3 + 2] = lz;
    }
    __syncthreads();
    for (int it = 1; it < NPROP; ++it) {
        float bestv = -1.0f; int besti = 0;
        for (int p = t; p < K_; p += 1024) {
            float dx = X[p * 3 + 0] - lx;
            float dy = X[p * 3 + 1] - ly;
            float dz = X[p * 3 + 2] - lz;
            float d2 = dx * dx + dy * dy + dz * dz;
            float dd = fminf(dist[p], d2);
            dist[p] = dd;
            if (dd > bestv || (dd == bestv && p < besti)) { bestv = dd; besti = p; }
        }
        for (int off = 16; off; off >>= 1) {      // wave32 argmax (ties -> lowest idx)
            float ov = __shfl_xor(bestv, off, 32);
            int   oi = __shfl_xor(besti, off, 32);
            if (ov > bestv || (ov == bestv && oi < besti)) { bestv = ov; besti = oi; }
        }
        if (lane == 0) { rv[w] = bestv; ri[w] = besti; }
        __syncthreads();
        if (w == 0) {
            float v = rv[lane]; int i2 = ri[lane];
            for (int off = 16; off; off >>= 1) {
                float ov = __shfl_xor(v, off, 32);
                int   oi = __shfl_xor(i2, off, 32);
                if (ov > v || (ov == v && oi < i2)) { v = ov; i2 = oi; }
            }
            if (lane == 0) bidx = i2;
        }
        __syncthreads();
        int last = bidx;
        lx = X[last * 3 + 0]; ly = X[last * 3 + 1]; lz = X[last * 3 + 2];
        if (t == 0) {
            fpsidx[b * NPROP + it] = last;
            newxyz[(b * NPROP + it) * 3 + 0] = lx;
            newxyz[(b * NPROP + it) * 3 + 1] = ly;
            newxyz[(b * NPROP + it) * 3 + 2] = lz;
        }
        __syncthreads();
    }
}

// ---------------------------------------------------------------------------
// Kernel 2: ball query, wave32 per proposal, ballot for in-order collection.
__global__ __launch_bounds__(256) void ballq_kernel(const float* __restrict__ xyz,
                                                    const float* __restrict__ newxyz,
                                                    int* ballidx) {
    int gw = (blockIdx.x * blockDim.x + threadIdx.x) >> 5;
    int lane = threadIdx.x & 31;
    if (gw >= B_ * NPROP) return;
    int b = gw / NPROP;
    const float* X = xyz + (size_t)b * K_ * 3;
    float cx = newxyz[gw * 3 + 0], cy = newxyz[gw * 3 + 1], cz = newxyz[gw * 3 + 2];
    int* out = ballidx + gw * NSAMP;
    const float r2 = RADIUS_ * RADIUS_;
    int cnt = 0, first = 0;
    for (int base = 0; base < K_ && cnt < NSAMP; base += 32) {
        int p = base + lane;
        float dx = X[p * 3 + 0] - cx;
        float dy = X[p * 3 + 1] - cy;
        float dz = X[p * 3 + 2] - cz;
        bool in = (dx * dx + dy * dy + dz * dz) < r2;
        unsigned mm = (unsigned)__ballot(in);        // wave32: low 32 bits
        while (mm && cnt < NSAMP) {
            int bit = __builtin_ctz(mm); mm &= mm - 1;
            int pi = base + bit;
            if (cnt == 0) first = pi;
            if (lane == 0) out[cnt] = pi;
            ++cnt;
        }
    }
    if (lane == 0) for (int i = cnt; i < NSAMP; ++i) out[i] = first;
}

// ---------------------------------------------------------------------------
// Kernel 3: fused gather -> 3x (WMMA GEMM + BN + ReLU) -> maxpool over samples.
// One workgroup (8 waves) per proposal: N-tile = 16 cols = this proposal's samples.
__global__ __launch_bounds__(256) void pn_mlp_kernel(const float* __restrict__ xyz,
                                                     const float* __restrict__ features,
                                                     const float* __restrict__ newxyz,
                                                     const int* __restrict__ ballidx,
                                                     const unsigned short* W1s,
                                                     const unsigned short* W2s,
                                                     const unsigned short* W3s,
                                                     const float* scale, const float* shift,
                                                     unsigned short* featB) {
    int gp = blockIdx.x;                 // 0..4095 (b*512+n)
    int b = gp >> 9;
    __shared__ __align__(32) unsigned short G[KB1 * 512];   // swizzled B layout
    __shared__ __align__(32) unsigned short H1[KBC * 512];
    __shared__ __align__(32) unsigned short H2[KBC * 512];
    __shared__ int   pidx[16];
    __shared__ float ctr[3];
    int t = threadIdx.x;
    if (t < 16) pidx[t] = ballidx[gp * 16 + t];
    if (t < 3)  ctr[t]  = newxyz[gp * 3 + t];
    __syncthreads();
    const float* X = xyz + (size_t)b * K_ * 3;
    const float* F = features + (size_t)b * C_ * K_;
    for (int e = t; e < K1PAD * 16; e += 256) {
        int r = e >> 4, s = e & 15;
        float v;
        if (r < 3)        v = (X[pidx[s] * 3 + r] - ctr[r]) * (1.0f / RADIUS_);
        else if (r < 259) v = F[(r - 3) * K_ + pidx[s]];
        else              v = 0.0f;
        G[swz(r, s)] = f2bf(v);
    }
    __syncthreads();
    int lane = t & 31, w = t >> 5;
    int sub = lane >> 4, rr = lane & 15;

    v8f acc = {0, 0, 0, 0, 0, 0, 0, 0};
#pragma unroll
    for (int kk = 0; kk < KB1; ++kk) {
        v16bf a  = load_a(W1s, KB1, w, kk, lane);
        v16bf bb = load_b(G, kk, lane);
        acc = __builtin_amdgcn_wmma_f32_16x16x32_bf16(false, a, false, bb, (short)0, acc, false, false);
    }
    bn_relu_store(acc, 0, w, sub, rr, scale, shift, H1);
    __syncthreads();

    acc = (v8f){0, 0, 0, 0, 0, 0, 0, 0};
#pragma unroll
    for (int kk = 0; kk < KBC; ++kk) {
        v16bf a  = load_a(W2s, KBC, w, kk, lane);
        v16bf bb = load_b(H1, kk, lane);
        acc = __builtin_amdgcn_wmma_f32_16x16x32_bf16(false, a, false, bb, (short)0, acc, false, false);
    }
    bn_relu_store(acc, 1, w, sub, rr, scale, shift, H2);
    __syncthreads();

    acc = (v8f){0, 0, 0, 0, 0, 0, 0, 0};
#pragma unroll
    for (int kk = 0; kk < KBC; ++kk) {
        v16bf a  = load_a(W3s, KBC, w, kk, lane);
        v16bf bb = load_b(H2, kk, lane);
        acc = __builtin_amdgcn_wmma_f32_16x16x32_bf16(false, a, false, bb, (short)0, acc, false, false);
    }
    // BN+ReLU then max over the 16 samples (cross-lane within each 16-lane half)
#pragma unroll
    for (int j = 0; j < 8; ++j) {
        int ch = w * 16 + sub * 8 + j;
        float y = acc[j] * scale[2 * NCH + ch] + shift[2 * NCH + ch];
        y = fmaxf(y, 0.0f);
        for (int off = 8; off; off >>= 1) y = fmaxf(y, __shfl_xor(y, off, 32));
        if (rr == 0) featB[ch * NCOL + gp] = f2bf(y);
    }
}

// ---------------------------------------------------------------------------
// Kernel 4: fused FC chain (two 128x128 WMMA GEMMs + padded 119x128) + head.
__global__ __launch_bounds__(256) void fc_head_kernel(const unsigned short* featB,
                                                      const unsigned short* WC1s,
                                                      const unsigned short* WC2s,
                                                      const unsigned short* WC3s,
                                                      const float* scale, const float* shift,
                                                      const float* bc3f,
                                                      const float* __restrict__ newxyz,
                                                      const float* __restrict__ meansize,
                                                      float* out) {
    int col0 = blockIdx.x * 16;          // 256 blocks x 16 cols = 4096
    __shared__ __align__(32) unsigned short Fb[KBC * 512];
    __shared__ __align__(32) unsigned short H1[KBC * 512];
    __shared__ __align__(32) unsigned short H2[KBC * 512];
    int t = threadIdx.x;
    for (int e = t; e < NCH * 16; e += 256) {
        int r = e >> 4, c = e & 15;
        Fb[swz(r, c)] = featB[r * NCOL + col0 + c];
    }
    __syncthreads();
    int lane = t & 31, w = t >> 5;
    int sub = lane >> 4, rr = lane & 15;

    v8f acc = {0, 0, 0, 0, 0, 0, 0, 0};
#pragma unroll
    for (int kk = 0; kk < KBC; ++kk) {
        v16bf a  = load_a(WC1s, KBC, w, kk, lane);
        v16bf bb = load_b(Fb, kk, lane);
        acc = __builtin_amdgcn_wmma_f32_16x16x32_bf16(false, a, false, bb, (short)0, acc, false, false);
    }
    bn_relu_store(acc, 3, w, sub, rr, scale, shift, H1);
    __syncthreads();

    acc = (v8f){0, 0, 0, 0, 0, 0, 0, 0};
#pragma unroll
    for (int kk = 0; kk < KBC; ++kk) {
        v16bf a  = load_a(WC2s, KBC, w, kk, lane);
        v16bf bb = load_b(H1, kk, lane);
        acc = __builtin_amdgcn_wmma_f32_16x16x32_bf16(false, a, false, bb, (short)0, acc, false, false);
    }
    bn_relu_store(acc, 4, w, sub, rr, scale, shift, H2);
    __syncthreads();

    acc = (v8f){0, 0, 0, 0, 0, 0, 0, 0};
#pragma unroll
    for (int kk = 0; kk < KBC; ++kk) {
        v16bf a  = load_a(WC3s, KBC, w, kk, lane);
        v16bf bb = load_b(H2, kk, lane);
        acc = __builtin_amdgcn_wmma_f32_16x16x32_bf16(false, a, false, bb, (short)0, acc, false, false);
    }
    // Head: per-(channel,column) elementwise transform on the D tile.
    int col = col0 + rr;                   // col = b*512 + n
    const float PI_NH = 3.14159265358979323846f / 12.0f;
#pragma unroll
    for (int j = 0; j < 8; ++j) {
        int ch = w * 16 + sub * 8 + j;
        if (ch < OUTC) {
            float x = acc[j] + bc3f[ch];
            float y;
            if (ch >= 2 && ch < 5)        y = x + newxyz[col * 3 + (ch - 2)];
            else if (ch >= 17 && ch < 29) y = x * PI_NH;
            else if (ch >= 47 && ch < 101) {
                float sp = fmaxf(x, 0.0f) + log1pf(expf(-fabsf(x)));   // stable softplus
                y = (sp - 1.0f) * meansize[ch - 47];
            } else                         y = x;
            out[(size_t)col * OUTC + ch] = y;
        }
    }
}

// ---------------------------------------------------------------------------
extern "C" void kernel_launch(void* const* d_in, const int* in_sizes, int n_in,
                              void* d_out, int out_size, void* d_ws, size_t ws_size,
                              hipStream_t stream) {
    const float* xyz      = (const float*)d_in[0];
    const float* features = (const float*)d_in[1];
    const float* w1  = (const float*)d_in[2];
    const float* w2  = (const float*)d_in[3];
    const float* w3  = (const float*)d_in[4];
    const float* wc1 = (const float*)d_in[5];
    const float* bc1 = (const float*)d_in[6];
    const float* wc2 = (const float*)d_in[7];
    const float* bc2 = (const float*)d_in[8];
    const float* wc3 = (const float*)d_in[9];
    const float* bc3 = (const float*)d_in[10];
    const float* bng = (const float*)d_in[11];
    const float* bnb = (const float*)d_in[12];
    const float* bnm = (const float*)d_in[13];
    const float* bnv = (const float*)d_in[14];
    const float* msz = (const float*)d_in[15];
    float* out = (float*)d_out;

    char* ws = (char*)d_ws;
    int*            fpsidx  = (int*)(ws + 0);                       // 16384 B
    float*          newxyz  = (float*)(ws + 16384);                 // 49152 B
    int*            ballidx = (int*)(ws + 65536);                   // 262144 B
    unsigned short* W1s     = (unsigned short*)(ws + 327680);       // 73728 B
    unsigned short* W2s     = (unsigned short*)(ws + 401408);       // 32768 B
    unsigned short* W3s     = (unsigned short*)(ws + 434176);       // 32768 B
    unsigned short* WC1s    = (unsigned short*)(ws + 466944);       // 32768 B
    unsigned short* WC2s    = (unsigned short*)(ws + 499712);       // 32768 B
    unsigned short* WC3s    = (unsigned short*)(ws + 532480);       // 32768 B
    float*          scale   = (float*)(ws + 565248);                // 2560 B
    float*          shift   = (float*)(ws + 567808);                // 2560 B
    float*          bc3f    = (float*)(ws + 570368);                // 512 B
    unsigned short* featB   = (unsigned short*)(ws + 570880);       // 1048576 B

    prep_kernel<<<467, 256, 0, stream>>>(w1, w2, w3, wc1, bc1, wc2, bc2, wc3, bc3,
                                         bng, bnb, bnm, bnv,
                                         W1s, W2s, W3s, WC1s, WC2s, WC3s,
                                         scale, shift, bc3f);
    fps_kernel<<<B_, 1024, 0, stream>>>(xyz, fpsidx, newxyz);
    ballq_kernel<<<(B_ * NPROP * 32) / 256, 256, 0, stream>>>(xyz, newxyz, ballidx);
    pn_mlp_kernel<<<B_ * NPROP, 256, 0, stream>>>(xyz, features, newxyz, ballidx,
                                                  W1s, W2s, W3s, scale, shift, featB);
    fc_head_kernel<<<NCOL / 16, 256, 0, stream>>>(featB, WC1s, WC2s, WC3s,
                                                  scale, shift, bc3f, newxyz, msz, out);
}
